// GAT_83674552861190
// MI455X (gfx1250) — compile-verified, matrix operand order checked by reading, not run on hardware
//
#include <hip/hip_runtime.h>
#include <math.h>

// Problem constants (match reference)
#define NNODES 20000
#define NEDGES 320000
#define NHEADS 8
#define NCH 32
#define HCDIM 256
#define DINPUT 16
#define DEDGE 8
#define DLIN 16
#define DOUT 4
#define NETOT (NEDGES + NNODES)

typedef __attribute__((ext_vector_type(16))) _Float16 v16h;
typedef __attribute__((ext_vector_type(8)))  _Float16 v8h;
typedef __attribute__((ext_vector_type(8)))  float    v8f;

static inline int cdiv_i(int a, int b) { return (a + b - 1) / b; }

// ---------------------------------------------------------------- utilities
__global__ void fill_f32(float* __restrict__ p, float v, int n) {
    int i = blockIdx.x * blockDim.x + threadIdx.x;
    if (i < n) p[i] = v;
}

__global__ void copy_f32(float* __restrict__ dst, const float* __restrict__ src, int n) {
    int i = blockIdx.x * blockDim.x + threadIdx.x;
    if (i < n) dst[i] = src[i];
}

// ------------------------------------------------ self-loop edge_attr mean
__global__ void loop_accum(const int* __restrict__ dst, const float* __restrict__ eattr,
                           float* __restrict__ ea_loop, float* __restrict__ deg) {
    int e = blockIdx.x * blockDim.x + threadIdx.x;
    if (e >= NEDGES) return;
    int d = dst[e];
    atomicAdd(&deg[d], 1.0f);
#pragma unroll
    for (int j = 0; j < DEDGE; ++j)
        atomicAdd(&ea_loop[(size_t)d * DEDGE + j], eattr[(size_t)e * DEDGE + j]);
}

__global__ void loop_div(float* __restrict__ ea_loop, const float* __restrict__ deg) {
    int i = blockIdx.x * blockDim.x + threadIdx.x;
    if (i >= NNODES * DEDGE) return;
    ea_loop[i] /= fmaxf(deg[i / DEDGE], 1.0f);
}

// ------------------------------------------------------------ f16 staging
// activations: [rows x kin] f32 -> [rows x kpad] f16 (zero-padded K)
__global__ void cast_act(const float* __restrict__ a, _Float16* __restrict__ o,
                         int kin, int kpad) {
    int i = blockIdx.x * blockDim.x + threadIdx.x;
    if (i >= NNODES * kpad) return;
    int r = i / kpad, k = i - r * kpad;
    o[i] = (k < kin) ? (_Float16)a[(size_t)r * kin + k] : (_Float16)0.0f;
}

// weights: [kin x 256] f32 row-major -> transposed [256 x kpad] f16
__global__ void cast_wt(const float* __restrict__ w, _Float16* __restrict__ o,
                        int kin, int kpad) {
    int i = blockIdx.x * blockDim.x + threadIdx.x;
    if (i >= HCDIM * kpad) return;
    int n = i / kpad, k = i - n * kpad;
    o[i] = (k < kin) ? (_Float16)w[(size_t)k * HCDIM + n] : (_Float16)0.0f;
}

// --------------------------------------------------------- WMMA GEMM
// C[M x 256] = A[M x K] * W + bias, with Bt = W^T stored [256 x K] f16.
// grid = (M/16), block = 256 (8 waves). Each wave owns TWO adjacent 16x16
// output tiles sharing one A fragment -> two independent WMMA accumulator
// chains (2x ILP on the matrix pipe) and half the A-fragment traffic.
// Fragment layout per CDNA5 ISA 7.12.2 (16-bit A 16x32): lane&15 = row,
// lane>>4 selects K-group (+8); v16h elems 0..7 = K[kb..kb+7], 8..15 = K[kb+16..kb+23].
__global__ __launch_bounds__(256) void gemm_f16_wmma(
    const _Float16* __restrict__ A, const _Float16* __restrict__ Bt,
    const float* __restrict__ bias, float* __restrict__ C, int K) {
    const int lane = threadIdx.x & 31;
    const int wave = threadIdx.x >> 5;
    const int m0 = blockIdx.x * 16;
    const int n0 = wave * 32;                 // two 16-col tiles per wave
    const int half = lane >> 4;               // 0 or 1
    const int idx  = lane & 15;
    const _Float16* aptr  = A  + (size_t)(m0 + idx) * K + half * 8;
    const _Float16* b0ptr = Bt + (size_t)(n0 + idx) * K + half * 8;
    const _Float16* b1ptr = Bt + (size_t)(n0 + 16 + idx) * K + half * 8;

    v8f acc0 = {}, acc1 = {};
    for (int kk = 0; kk < K; kk += 32) {
        v8h al  = *(const v8h*)(aptr + kk);
        v8h ah  = *(const v8h*)(aptr + kk + 16);
        v8h b0l = *(const v8h*)(b0ptr + kk);
        v8h b0h = *(const v8h*)(b0ptr + kk + 16);
        v8h b1l = *(const v8h*)(b1ptr + kk);
        v8h b1h = *(const v8h*)(b1ptr + kk + 16);
        v16h a  = __builtin_shufflevector(al,  ah,  0,1,2,3,4,5,6,7,8,9,10,11,12,13,14,15);
        v16h b0 = __builtin_shufflevector(b0l, b0h, 0,1,2,3,4,5,6,7,8,9,10,11,12,13,14,15);
        v16h b1 = __builtin_shufflevector(b1l, b1h, 0,1,2,3,4,5,6,7,8,9,10,11,12,13,14,15);
        acc0 = __builtin_amdgcn_wmma_f32_16x16x32_f16(
            false, a, false, b0, (short)0, acc0, false, false);
        acc1 = __builtin_amdgcn_wmma_f32_16x16x32_f16(
            false, a, false, b1, (short)0, acc1, false, false);
    }
    const int col0 = n0 + idx;
    const int col1 = n0 + 16 + idx;
    const float bv0 = bias[col0];
    const float bv1 = bias[col1];
#pragma unroll
    for (int r = 0; r < 8; ++r) {
        int row = m0 + half * 8 + r;          // C layout: VGPR r -> M = r + 8*(lane>=16)
        C[(size_t)row * HCDIM + col0] = acc0[r] + bv0;
        C[(size_t)row * HCDIM + col1] = acc1[r] + bv1;
    }
}

// ------------------------------------------------------------ edge passes
__device__ inline void atomicMaxF(float* addr, float val) {
    int* ia = (int*)addr;
    int old = *ia;
    while (__int_as_float(old) < val) {
        int assumed = old;
        old = atomicCAS(ia, assumed, __float_as_int(val));
        if (old == assumed) break;
    }
}

// pass 1: score[e,h] = sum_c leaky(xl[s]+xr[d]+ea@We) * att ; smax = segmax
__global__ void edge_score(const float* __restrict__ ea, const int* __restrict__ src,
                           const int* __restrict__ dst, const float* __restrict__ We,
                           const float* __restrict__ att, const float* __restrict__ xl,
                           const float* __restrict__ xr, float* __restrict__ score,
                           float* __restrict__ smax) {
    int t = blockIdx.x * blockDim.x + threadIdx.x;
    if (t >= NETOT * NHEADS) return;
    int e = t >> 3, h = t & 7;
    int s = (e < NEDGES) ? src[e] : (e - NEDGES);
    int d = (e < NEDGES) ? dst[e] : (e - NEDGES);
    float eav[DEDGE];
#pragma unroll
    for (int j = 0; j < DEDGE; ++j) eav[j] = ea[(size_t)e * DEDGE + j];
    const float* xlp  = xl  + (size_t)s * HCDIM + h * NCH;
    const float* xrp  = xr  + (size_t)d * HCDIM + h * NCH;
    const float* attp = att + h * NCH;
    const float* wep  = We  + h * NCH;
    float sc = 0.0f;
#pragma unroll 4
    for (int c = 0; c < NCH; ++c) {
        float ee = 0.0f;
#pragma unroll
        for (int j = 0; j < DEDGE; ++j) ee += eav[j] * wep[j * HCDIM + c];
        float m = xlp[c] + xrp[c] + ee;
        m = (m > 0.0f) ? m : 0.2f * m;
        sc += m * attp[c];
    }
    score[t] = sc;
    atomicMaxF(&smax[d * NHEADS + h], sc);
}

// pass 2: score <- exp(score - smax[d]); denom = segsum
__global__ void edge_exp(const int* __restrict__ dst, float* __restrict__ score,
                         const float* __restrict__ smax, float* __restrict__ denom) {
    int t = blockIdx.x * blockDim.x + threadIdx.x;
    if (t >= NETOT * NHEADS) return;
    int e = t >> 3, h = t & 7;
    int d = (e < NEDGES) ? dst[e] : (e - NEDGES);
    float ex = __expf(score[t] - smax[d * NHEADS + h]);
    score[t] = ex;
    atomicAdd(&denom[d * NHEADS + h], ex);
}

// pass 3: out[d] += (ex/denom[d]) * xl[s]
__global__ void edge_aggr(const int* __restrict__ src, const int* __restrict__ dst,
                          const float* __restrict__ score, const float* __restrict__ denom,
                          const float* __restrict__ xl, float* __restrict__ out) {
    int t = blockIdx.x * blockDim.x + threadIdx.x;
    if (t >= NETOT * NHEADS) return;
    int e = t >> 3, h = t & 7;
    int s = (e < NEDGES) ? src[e] : (e - NEDGES);
    int d = (e < NEDGES) ? dst[e] : (e - NEDGES);
    float alpha = score[t] / denom[d * NHEADS + h];
    const float* xlp = xl + (size_t)s * HCDIM + h * NCH;
    float* op = out + (size_t)d * HCDIM + h * NCH;
#pragma unroll
    for (int c = 0; c < NCH; ++c) atomicAdd(&op[c], alpha * xlp[c]);
}

// h = leaky_relu(h + bc)
__global__ void bias_leaky(float* __restrict__ h, const float* __restrict__ b, int n) {
    int i = blockIdx.x * blockDim.x + threadIdx.x;
    if (i >= n) return;
    float v = h[i] + b[i & (HCDIM - 1)];
    h[i] = (v > 0.0f) ? v : 0.2f * v;
}

// --------------------------------------------------------------- MLP head
__global__ void mlp_head(const float* __restrict__ h,
                         const float* __restrict__ W0, const float* __restrict__ b0,
                         const float* __restrict__ W1, const float* __restrict__ b1,
                         const float* __restrict__ W2, const float* __restrict__ b2,
                         const float* __restrict__ W3, const float* __restrict__ b3,
                         float* __restrict__ out) {
    int n = blockIdx.x * blockDim.x + threadIdx.x;
    if (n >= NNODES) return;
    float a[DLIN], t[DLIN];
#pragma unroll
    for (int j = 0; j < DLIN; ++j) a[j] = b0[j];
    for (int k = 0; k < HCDIM; ++k) {
        float hv = h[(size_t)n * HCDIM + k];
#pragma unroll
        for (int j = 0; j < DLIN; ++j) a[j] += hv * W0[k * DLIN + j];
    }
#pragma unroll
    for (int j = 0; j < DLIN; ++j) a[j] = fmaxf(a[j], 0.0f);

#pragma unroll
    for (int j = 0; j < DLIN; ++j) t[j] = b1[j];
    for (int k = 0; k < DLIN; ++k)
#pragma unroll
        for (int j = 0; j < DLIN; ++j) t[j] += a[k] * W1[k * DLIN + j];
#pragma unroll
    for (int j = 0; j < DLIN; ++j) a[j] = fmaxf(t[j], 0.0f);

#pragma unroll
    for (int j = 0; j < DLIN; ++j) t[j] = b2[j];
    for (int k = 0; k < DLIN; ++k)
#pragma unroll
        for (int j = 0; j < DLIN; ++j) t[j] += a[k] * W2[k * DLIN + j];
#pragma unroll
    for (int j = 0; j < DLIN; ++j) a[j] = fmaxf(t[j], 0.0f);

    float o[DOUT];
#pragma unroll
    for (int j = 0; j < DOUT; ++j) o[j] = b3[j];
    for (int k = 0; k < DLIN; ++k)
#pragma unroll
        for (int j = 0; j < DOUT; ++j) o[j] += a[k] * W3[k * DOUT + j];
#pragma unroll
    for (int j = 0; j < DOUT; ++j) out[(size_t)n * DOUT + j] = o[j];
}

// -------------------------------------------------------------- launcher
extern "C" void kernel_launch(void* const* d_in, const int* in_sizes, int n_in,
                              void* d_out, int out_size, void* d_ws, size_t ws_size,
                              hipStream_t stream) {
    (void)in_sizes; (void)n_in; (void)out_size; (void)ws_size;

    const float* x     = (const float*)d_in[0];
    const int*   ei    = (const int*)d_in[1];
    const int*   srcp  = ei;
    const int*   dstp  = ei + NEDGES;
    const float* eattr = (const float*)d_in[2];

    const float *Wl[3], *bl[3], *Wr[3], *br[3], *We[3], *att[3], *bc[3];
    for (int l = 0; l < 3; ++l) {
        int b = 3 + l * 7;
        Wl[l]  = (const float*)d_in[b + 0];
        bl[l]  = (const float*)d_in[b + 1];
        Wr[l]  = (const float*)d_in[b + 2];
        br[l]  = (const float*)d_in[b + 3];
        We[l]  = (const float*)d_in[b + 4];
        att[l] = (const float*)d_in[b + 5];
        bc[l]  = (const float*)d_in[b + 6];
    }
    const float* lW0 = (const float*)d_in[24]; const float* lb0 = (const float*)d_in[25];
    const float* lW1 = (const float*)d_in[26]; const float* lb1 = (const float*)d_in[27];
    const float* lW2 = (const float*)d_in[28]; const float* lb2 = (const float*)d_in[29];
    const float* lW3 = (const float*)d_in[30]; const float* lb3 = (const float*)d_in[31];

    // workspace carve (256-B aligned); ~116 MB total, L2-resident working set
    char* wp = (char*)d_ws;
    auto carve = [&](size_t bytes) -> char* {
        char* p = wp;
        wp += (bytes + 255) & ~(size_t)255;
        return p;
    };
    float*    ea    = (float*)carve((size_t)NETOT * DEDGE * 4);
    float*    deg   = (float*)carve((size_t)NNODES * 4);
    _Float16* h16   = (_Float16*)carve((size_t)NNODES * HCDIM * 2);
    _Float16* wl16  = (_Float16*)carve((size_t)HCDIM * HCDIM * 2);
    _Float16* wr16  = (_Float16*)carve((size_t)HCDIM * HCDIM * 2);
    float*    xl    = (float*)carve((size_t)NNODES * HCDIM * 4);
    float*    xr    = (float*)carve((size_t)NNODES * HCDIM * 4);
    float*    score = (float*)carve((size_t)NETOT * NHEADS * 4);
    float*    smax  = (float*)carve((size_t)NNODES * NHEADS * 4);
    float*    denom = (float*)carve((size_t)NNODES * NHEADS * 4);
    float*    hA    = (float*)carve((size_t)NNODES * HCDIM * 4);
    float*    hB    = (float*)carve((size_t)NNODES * HCDIM * 4);

    const int T = 256;
    float* ea_loop = ea + (size_t)NEDGES * DEDGE;

    // self-loop edge_attr = per-dst mean of incoming edge_attr (layer-invariant)
    fill_f32<<<cdiv_i(NNODES, T), T, 0, stream>>>(deg, 0.0f, NNODES);
    fill_f32<<<cdiv_i(NNODES * DEDGE, T), T, 0, stream>>>(ea_loop, 0.0f, NNODES * DEDGE);
    copy_f32<<<cdiv_i(NEDGES * DEDGE, T), T, 0, stream>>>(ea, eattr, NEDGES * DEDGE);
    loop_accum<<<cdiv_i(NEDGES, T), T, 0, stream>>>(dstp, eattr, ea_loop, deg);
    loop_div<<<cdiv_i(NNODES * DEDGE, T), T, 0, stream>>>(ea_loop, deg);

    const float* hin = x;
    float* houts[3] = { hA, hB, hA };
    int kins[3] = { DINPUT, HCDIM, HCDIM };

    for (int l = 0; l < 3; ++l) {
        int kin = kins[l];
        int kpad = (kin + 31) & ~31;      // 32 for layer 0, 256 otherwise

        cast_act<<<cdiv_i(NNODES * kpad, T), T, 0, stream>>>(hin, h16, kin, kpad);
        cast_wt<<<cdiv_i(HCDIM * kpad, T), T, 0, stream>>>(Wl[l], wl16, kin, kpad);
        cast_wt<<<cdiv_i(HCDIM * kpad, T), T, 0, stream>>>(Wr[l], wr16, kin, kpad);

        dim3 gg(NNODES / 16);             // 1250 blocks; 8 waves x two 16x16 tiles = 256 cols
        gemm_f16_wmma<<<gg, 256, 0, stream>>>(h16, wl16, bl[l], xl, kpad);
        gemm_f16_wmma<<<gg, 256, 0, stream>>>(h16, wr16, br[l], xr, kpad);

        float* hout = houts[l];
        fill_f32<<<cdiv_i(NNODES * NHEADS, T), T, 0, stream>>>(smax, -INFINITY, NNODES * NHEADS);
        fill_f32<<<cdiv_i(NNODES * NHEADS, T), T, 0, stream>>>(denom, 0.0f, NNODES * NHEADS);
        fill_f32<<<cdiv_i(NNODES * HCDIM, T), T, 0, stream>>>(hout, 0.0f, NNODES * HCDIM);

        int eh = NETOT * NHEADS;
        edge_score<<<cdiv_i(eh, T), T, 0, stream>>>(ea, srcp, dstp, We[l], att[l], xl, xr, score, smax);
        edge_exp<<<cdiv_i(eh, T), T, 0, stream>>>(dstp, score, smax, denom);
        edge_aggr<<<cdiv_i(eh, T), T, 0, stream>>>(srcp, dstp, score, denom, xl, hout);
        bias_leaky<<<cdiv_i(NNODES * HCDIM, T), T, 0, stream>>>(hout, bc[l], NNODES * HCDIM);

        hin = hout;
    }

    mlp_head<<<cdiv_i(NNODES, T), T, 0, stream>>>(hA, lW0, lb0, lW1, lb1, lW2, lb2, lW3, lb3,
                                                  (float*)d_out);
}
